// DispAttentionBlock_44693429682780
// MI455X (gfx1250) — matrix operand order repaired and checked by reading, not compiled, not run on hardware
//
#include <hip/hip_runtime.h>

#define HN 8
#define HS 32
#define EMB 256
#define WW 128
#define HH 256
#define BB 4
#define HW (HH*WW)          // 32768
#define FF 1024

typedef __attribute__((ext_vector_type(16))) __bf16 bf16x16;
typedef __attribute__((ext_vector_type(8)))  float  f32x8;
typedef __attribute__((ext_vector_type(4)))  unsigned int u32x4;
typedef __attribute__((ext_vector_type(8)))  int i32x8;
typedef __attribute__((ext_vector_type(4)))  int i32x4;

// ---------- bf16 <-> f32 bit helpers (RNE) ----------
__device__ __forceinline__ float bf2f(__bf16 v) {
    union { __bf16 b; unsigned short u; } c; c.b = v;
    unsigned int x = ((unsigned int)c.u) << 16;
    float f; __builtin_memcpy(&f, &x, 4); return f;
}
__device__ __forceinline__ __bf16 f2bf(float f) {
    unsigned int x; __builtin_memcpy(&x, &f, 4);
    unsigned int r = x + 0x7FFFu + ((x >> 16) & 1u);
    unsigned short u = (unsigned short)(r >> 16);
    union { unsigned short u; __bf16 b; } c; c.u = u; return c.b;
}
__device__ __forceinline__ unsigned lds_offset(const void* p) {
    // generic LDS pointer = {shared_aperture_hi, 32'lds_offset}; low 32 bits = offset
    return (unsigned)(size_t)p;
}

// ---------- Tensor Data Mover: 2D tile load, global -> LDS ----------
// D# per CDNA5 ISA ch.8. ROWS x DPR dwords, source row stride STRIDE_DW dwords.
// pad: 1 dword inserted after every 16 dwords -> LDS row stride = 17 dwords
// (== 34 bf16), matching the WMMA fragment loaders' bank-conflict-free layout.
// 6-arg builtin form (clang-23 / amdgpu-toolchain): (g0, g1, g2, g3, g4, cpol)
template<int ROWS, int DPR, int STRIDE_DW>
__device__ __forceinline__ void tdm_load_2d(unsigned lds_off, const void* gptr) {
    unsigned long long ga = (unsigned long long)(size_t)gptr;
    u32x4 g0 = { 1u,                                   // count=1 (valid), user mode
                 lds_off,                              // lds_addr
                 (unsigned)ga,                         // global_addr[31:0]
                 (unsigned)((ga >> 32) & 0x1FFFFFFu) | 0x80000000u }; // [56:32] | type=2
    i32x8 g1 = { (int)((2u << 16) | (1u << 20) | (3u << 22)), // data_size=4B, pad_en, interval=16dw (amount=1dw)
                 0,                                    // atomic_barrier=0, tensor_dim0 lo16=0
                 0x4000,                               // tensor_dim0=0x40000000 (no OOB), dim1 lo16=0
                 (int)(0x4000u | ((unsigned)DPR << 16)), // tensor_dim1=0x40000000, tile_dim0=DPR
                 ROWS,                                 // tile_dim1=ROWS, tile_dim2=0
                 STRIDE_DW,                            // tensor_dim0_stride[31:0]
                 0, 0 };
    i32x4 z4 = { 0, 0, 0, 0 };
    i32x8 z8 = { 0, 0, 0, 0, 0, 0, 0, 0 };
    __builtin_amdgcn_tensor_load_to_lds(g0, g1, z4, z4, z8, 0);
}

// ---------- WMMA fragment loaders (CDNA5 wave32 layouts, ISA 7.12.2) ----------
template<int STRIDE>
__device__ __forceinline__ bf16x16 load_a(const __bf16* X, int m0, int k0) {
    const int lane = threadIdx.x & 31;
    const int r = lane & 15, hf = lane >> 4;
    bf16x16 a;
#pragma unroll
    for (int i = 0; i < 8; ++i) {
        int k = (i < 4) ? (hf * 8 + 2 * i) : (16 + hf * 8 + 2 * (i - 4));
        const __bf16* p = X + (m0 + r) * STRIDE + k0 + k;
        a[2 * i] = p[0];
        a[2 * i + 1] = p[1];
    }
    return a;
}
template<int STRIDE>
__device__ __forceinline__ bf16x16 load_b(const __bf16* Bt, int n0, int k0) {
    const int lane = threadIdx.x & 31;
    const int c = lane & 15, hf = lane >> 4;
    bf16x16 b;
#pragma unroll
    for (int i = 0; i < 8; ++i) {
        int k = (i < 4) ? (hf * 8 + 2 * i) : (16 + hf * 8 + 2 * (i - 4));
        const __bf16* p = Bt + (n0 + c) * STRIDE + k0 + k;
        b[2 * i] = p[0];
        b[2 * i + 1] = p[1];
    }
    return b;
}
__device__ __forceinline__ f32x8 wmma_bf16(bf16x16 a, bf16x16 b, f32x8 c) {
    return __builtin_amdgcn_wmma_f32_16x16x32_bf16(false, a, false, b, (short)0, c,
                                                   false, false);
}

// =====================================================================
// Kernel A: grouped 1x1 projections (32x32 per head) + RoPE for q,k
// =====================================================================
__global__ __launch_bounds__(256)
void qkv_rope_kernel(const float* __restrict__ left, const float* __restrict__ right,
                     const float* __restrict__ qw, const float* __restrict__ kw,
                     const float* __restrict__ vw,
                     __bf16* __restrict__ qb, __bf16* __restrict__ kb,
                     __bf16* __restrict__ vb) {
    const int which = blockIdx.z;
    const int b = blockIdx.y / HN, g = blockIdx.y % HN;
    const int pos0 = blockIdx.x * 128;
    const float* src  = (which == 0) ? left : right;
    const float* wmat = (which == 0) ? qw : (which == 1 ? kw : vw);
    __bf16* dst       = (which == 0) ? qb : (which == 1 ? kb : vb);

    __shared__ __bf16 Xs[128 * 34];   // [pos][ch]
    __shared__ __bf16 Ws[32 * 34];    // [o][i] == Bt layout [N][K]
    const int tid = threadIdx.x;
#pragma unroll
    for (int it = 0; it < 16; ++it) {
        int idx = tid + it * 256;
        int ch = idx >> 7, p = idx & 127;
        Xs[p * 34 + ch] = f2bf(src[(size_t)(b * EMB + g * HS + ch) * HW + pos0 + p]);
    }
#pragma unroll
    for (int it = 0; it < 4; ++it) {
        int idx = tid + it * 256;
        int o = idx >> 5, i = idx & 31;
        Ws[o * 34 + i] = f2bf(wmat[(g * HS + o) * HS + i]);
    }
    __syncthreads();

    const int wave = tid >> 5, lane = tid & 31;
    const int nl = lane & 15, hf = lane >> 4;
    const int m0 = wave * 16;

    bf16x16 a = load_a<34>(Xs, m0, 0);
    f32x8 c0 = {}, c1 = {};
    c0 = wmma_bf16(a, load_b<34>(Ws, 0, 0), c0);
    c1 = wmma_bf16(a, load_b<34>(Ws, 16, 0), c1);

    const size_t base = ((size_t)(b * HN + g) * HW + pos0) * HS;
    if (which < 2) {
        float invf = __expf(-(float)nl * (4.605170185988091f / 16.0f)); // 100^(-nl/16)
#pragma unroll
        for (int r = 0; r < 8; ++r) {
            int row = m0 + r + 8 * hf;       // position within W
            float sn, cn;
            __sincosf((float)row * invf, &sn, &cn);
            float x0 = c0[r], x1 = c1[r];
            c0[r] = x0 * cn - x1 * sn;
            c1[r] = x1 * cn + x0 * sn;
        }
    }
#pragma unroll
    for (int r = 0; r < 8; ++r) {
        int row = m0 + r + 8 * hf;
        dst[base + (size_t)row * HS + nl]      = f2bf(c0[r]);
        dst[base + (size_t)row * HS + nl + 16] = f2bf(c1[r]);
    }
}

// =====================================================================
// Kernel B: attention: S = QK^T*scale, softpick, O = PV
// Q/K tiles staged by TDM (padded to stride 34); V transposed cooperatively
// =====================================================================
__global__ __launch_bounds__(256)
void attention_kernel(const __bf16* __restrict__ qb, const __bf16* __restrict__ kb,
                      const __bf16* __restrict__ vb, __bf16* __restrict__ ob) {
    const int head = blockIdx.x;
    const int g = head % HN;
    const int bh = head / HN;
    const int h = bh % HH, b = bh / HH;
    const size_t base = ((size_t)(b * HN + g) * HW + h * WW) * HS;

    __shared__ __bf16 Qs[128 * 34];
    __shared__ __bf16 Ks[128 * 34];
    __shared__ __bf16 Vt[32 * 130];
    __shared__ __bf16 Ps[128 * 130];

    const int tid = threadIdx.x;
    if (tid < 32) {  // wave 0 drives the Tensor Data Mover
        tdm_load_2d<128, 16, 16>(lds_offset(Qs), qb + base);
        tdm_load_2d<128, 16, 16>(lds_offset(Ks), kb + base);
        __builtin_amdgcn_s_wait_tensorcnt(0);
    }
#pragma unroll
    for (int it = 0; it < 16; ++it) {
        int idx = tid + it * 256;
        int p = idx >> 5, d = idx & 31;
        Vt[d * 130 + p] = vb[base + idx];
    }
    __syncthreads();

    const int wave = tid >> 5, lane = tid & 31;
    const int nl = lane & 15, hf = lane >> 4;
    const int m0 = wave * 16;

    f32x8 cs[8];
#pragma unroll
    for (int nb = 0; nb < 8; ++nb) cs[nb] = {};
    bf16x16 aq = load_a<34>(Qs, m0, 0);
#pragma unroll
    for (int nb = 0; nb < 8; ++nb)
        cs[nb] = wmma_bf16(aq, load_b<34>(Ks, nb * 16, 0), cs[nb]);

    const float scale = 0.17677669529663687f;  // 32^-0.5
#pragma unroll
    for (int r = 0; r < 8; ++r) {
        float mx = -3.0e38f;
#pragma unroll
        for (int nb = 0; nb < 8; ++nb) mx = fmaxf(mx, cs[nb][r] * scale);
        mx = fmaxf(mx, __shfl_xor(mx, 1));
        mx = fmaxf(mx, __shfl_xor(mx, 2));
        mx = fmaxf(mx, __shfl_xor(mx, 4));
        mx = fmaxf(mx, __shfl_xor(mx, 8));
        float em = __expf(-mx);
        float num[8];
        float den = 0.0f;
#pragma unroll
        for (int nb = 0; nb < 8; ++nb) {
            float e = __expf(cs[nb][r] * scale - mx) - em;
            num[nb] = fmaxf(e, 0.0f);
            den += fabsf(e);
        }
        den += __shfl_xor(den, 1);
        den += __shfl_xor(den, 2);
        den += __shfl_xor(den, 4);
        den += __shfl_xor(den, 8);
        float inv = 1.0f / (den + 1e-6f);
        int row = m0 + r + 8 * hf;
#pragma unroll
        for (int nb = 0; nb < 8; ++nb)
            Ps[row * 130 + nb * 16 + nl] = f2bf(num[nb] * inv);
    }
    __syncthreads();

    f32x8 d0 = {}, d1 = {};
#pragma unroll
    for (int kk = 0; kk < 4; ++kk) {
        bf16x16 ap = load_a<130>(Ps, m0, kk * 32);
        d0 = wmma_bf16(ap, load_b<130>(Vt, 0, kk * 32), d0);
        d1 = wmma_bf16(ap, load_b<130>(Vt, 16, kk * 32), d1);
    }
    const size_t obase = ((size_t)(b * HH + h) * WW) * EMB + g * HS;
#pragma unroll
    for (int r = 0; r < 8; ++r) {
        int row = m0 + r + 8 * hf;
        ob[obase + (size_t)row * EMB + nl]      = f2bf(d0[r]);
        ob[obase + (size_t)row * EMB + nl + 16] = f2bf(d1[r]);
    }
}

// =====================================================================
// Kernel C: x1 = LN(O @ proj_w + proj_b + left^T); M-tile 64, N=256 full row
// A tiles: TDM double-buffered; B tile: cooperative f32->bf16 convert
// =====================================================================
__global__ __launch_bounds__(256)
void proj_ln_kernel(const __bf16* __restrict__ ob, const float* __restrict__ pw,
                    const float* __restrict__ pb, const float* __restrict__ left,
                    const float* __restrict__ g1, const float* __restrict__ bb1,
                    __bf16* __restrict__ x1b) {
    __shared__ __align__(16) char smem[65536];
    __bf16* As0 = (__bf16*)smem;                   // 64*34 = 4352 B
    __bf16* As1 = (__bf16*)(smem + 4352);          // 64*34 = 4352 B
    __bf16* Bt  = (__bf16*)(smem + 8704);          // 256*34 = 17408 B
    float*  Rs  = (float*)smem;                    // 64*256 f32 (reused post-GEMM)

    const int tid = threadIdx.x;
    const int m_base = blockIdx.x * 64;
    const int wave = tid >> 5, lane = tid & 31;
    const int nl = lane & 15, hf = lane >> 4;
    const int wrow = wave >> 1, wcol = wave & 1;
    const int m0 = wrow * 16;

    if (tid < 32)
        tdm_load_2d<64, 16, 128>(lds_offset(As0), ob + (size_t)m_base * EMB);

    f32x8 cs[8];
#pragma unroll
    for (int j = 0; j < 8; ++j) cs[j] = {};

    for (int kk = 0; kk < 8; ++kk) {
#pragma unroll
        for (int it = 0; it < 32; ++it) {
            int idx = tid + it * 256;
            int kl = idx >> 8, n = idx & 255;
            Bt[n * 34 + kl] = f2bf(pw[(size_t)(kk * 32 + kl) * EMB + n]);
        }
        if (tid < 32) {
            if (kk + 1 < 8) {
                unsigned dstOff = lds_offset(((kk + 1) & 1) ? As1 : As0);
                tdm_load_2d<64, 16, 128>(dstOff, ob + (size_t)m_base * EMB + (kk + 1) * 32);
                __builtin_amdgcn_s_wait_tensorcnt(1);
            } else {
                __builtin_amdgcn_s_wait_tensorcnt(0);
            }
        }
        __syncthreads();
        const __bf16* As = (kk & 1) ? As1 : As0;
        bf16x16 a = load_a<34>(As, m0, 0);
#pragma unroll
        for (int j = 0; j < 8; ++j)
            cs[j] = wmma_bf16(a, load_b<34>(Bt, wcol * 128 + j * 16, 0), cs[j]);
        __syncthreads();
    }

#pragma unroll
    for (int j = 0; j < 8; ++j) {
        int col = wcol * 128 + j * 16 + nl;
#pragma unroll
        for (int r = 0; r < 8; ++r) {
            int row = m0 + r + 8 * hf;
            int mg = m_base + row;
            float resid = left[(size_t)((mg >> 15) * EMB + col) * HW + (mg & (HW - 1))];
            Rs[row * 256 + col] = cs[j][r] + pb[col] + resid;
        }
    }
    __syncthreads();

    for (int rr = wave * 8; rr < wave * 8 + 8; ++rr) {
        int mg = m_base + rr;
        float s = 0.0f;
#pragma unroll
        for (int jj = 0; jj < 8; ++jj) s += Rs[rr * 256 + lane + jj * 32];
        s += __shfl_xor(s, 1);  s += __shfl_xor(s, 2);  s += __shfl_xor(s, 4);
        s += __shfl_xor(s, 8);  s += __shfl_xor(s, 16);
        float mu = s * (1.0f / 256.0f);
        float vs = 0.0f;
#pragma unroll
        for (int jj = 0; jj < 8; ++jj) {
            float d = Rs[rr * 256 + lane + jj * 32] - mu;
            vs += d * d;
        }
        vs += __shfl_xor(vs, 1);  vs += __shfl_xor(vs, 2);  vs += __shfl_xor(vs, 4);
        vs += __shfl_xor(vs, 8);  vs += __shfl_xor(vs, 16);
        float inv = rsqrtf(vs * (1.0f / 256.0f) + 1e-5f);
#pragma unroll
        for (int jj = 0; jj < 8; ++jj) {
            int col = lane + jj * 32;
            float y = (Rs[rr * 256 + col] - mu) * inv * g1[col] + bb1[col];
            x1b[(size_t)mg * EMB + col] = f2bf(y);
        }
    }
}

// =====================================================================
// Kernel D: h = gelu_exact(x1 @ w1 + b1); grid (2048, 8)
// =====================================================================
__global__ __launch_bounds__(256)
void ffn1_kernel(const __bf16* __restrict__ x1b, const float* __restrict__ w1,
                 const float* __restrict__ b1f, __bf16* __restrict__ hbuf) {
    __shared__ __bf16 As0[64 * 34];
    __shared__ __bf16 As1[64 * 34];
    __shared__ __bf16 Bt[128 * 34];

    const int tid = threadIdx.x;
    const int m_base = blockIdx.x * 64;
    const int n_base = blockIdx.y * 128;
    const int wave = tid >> 5, lane = tid & 31;
    const int nl = lane & 15, hf = lane >> 4;
    const int wrow = wave >> 1, wcol = wave & 1;
    const int m0 = wrow * 16;

    if (tid < 32)
        tdm_load_2d<64, 16, 128>(lds_offset(As0), x1b + (size_t)m_base * EMB);

    f32x8 cs[4];
#pragma unroll
    for (int j = 0; j < 4; ++j) cs[j] = {};

    for (int kk = 0; kk < 8; ++kk) {
#pragma unroll
        for (int it = 0; it < 16; ++it) {
            int idx = tid + it * 256;
            int kl = idx >> 7, n = idx & 127;
            Bt[n * 34 + kl] = f2bf(w1[(size_t)(kk * 32 + kl) * FF + n_base + n]);
        }
        if (tid < 32) {
            if (kk + 1 < 8) {
                unsigned dstOff = lds_offset(((kk + 1) & 1) ? As1 : As0);
                tdm_load_2d<64, 16, 128>(dstOff, x1b + (size_t)m_base * EMB + (kk + 1) * 32);
                __builtin_amdgcn_s_wait_tensorcnt(1);
            } else {
                __builtin_amdgcn_s_wait_tensorcnt(0);
            }
        }
        __syncthreads();
        const __bf16* As = (kk & 1) ? As1 : As0;
        bf16x16 a = load_a<34>(As, m0, 0);
#pragma unroll
        for (int j = 0; j < 4; ++j)
            cs[j] = wmma_bf16(a, load_b<34>(Bt, wcol * 64 + j * 16, 0), cs[j]);
        __syncthreads();
    }

#pragma unroll
    for (int j = 0; j < 4; ++j) {
        int col = n_base + wcol * 64 + j * 16 + nl;
        float bias = b1f[col];
#pragma unroll
        for (int r = 0; r < 8; ++r) {
            int row = m0 + r + 8 * hf;
            float y = cs[j][r] + bias;
            float ge = 0.5f * y * (1.0f + erff(y * 0.7071067811865476f));
            hbuf[(size_t)(m_base + row) * FF + col] = f2bf(ge);
        }
    }
}

// =====================================================================
// Kernel E: out = LN(h @ w2 + b2 + x1); grid (2048)
// =====================================================================
__global__ __launch_bounds__(256)
void ffn2_ln_kernel(const __bf16* __restrict__ hbuf, const float* __restrict__ w2,
                    const float* __restrict__ b2f, const __bf16* __restrict__ x1b,
                    const float* __restrict__ g2, const float* __restrict__ bb2,
                    float* __restrict__ out) {
    __shared__ __align__(16) char smem[65536];
    __bf16* As0 = (__bf16*)smem;
    __bf16* As1 = (__bf16*)(smem + 4352);
    __bf16* Bt  = (__bf16*)(smem + 8704);
    float*  Rs  = (float*)smem;

    const int tid = threadIdx.x;
    const int m_base = blockIdx.x * 64;
    const int wave = tid >> 5, lane = tid & 31;
    const int nl = lane & 15, hf = lane >> 4;
    const int wrow = wave >> 1, wcol = wave & 1;
    const int m0 = wrow * 16;

    if (tid < 32)
        tdm_load_2d<64, 16, 512>(lds_offset(As0), hbuf + (size_t)m_base * FF);

    f32x8 cs[8];
#pragma unroll
    for (int j = 0; j < 8; ++j) cs[j] = {};

    for (int kk = 0; kk < 32; ++kk) {
#pragma unroll
        for (int it = 0; it < 32; ++it) {
            int idx = tid + it * 256;
            int kl = idx >> 8, n = idx & 255;
            Bt[n * 34 + kl] = f2bf(w2[(size_t)(kk * 32 + kl) * EMB + n]);
        }
        if (tid < 32) {
            if (kk + 1 < 32) {
                unsigned dstOff = lds_offset(((kk + 1) & 1) ? As1 : As0);
                tdm_load_2d<64, 16, 512>(dstOff, hbuf + (size_t)m_base * FF + (kk + 1) * 32);
                __builtin_amdgcn_s_wait_tensorcnt(1);
            } else {
                __builtin_amdgcn_s_wait_tensorcnt(0);
            }
        }
        __syncthreads();
        const __bf16* As = (kk & 1) ? As1 : As0;
        bf16x16 a = load_a<34>(As, m0, 0);
#pragma unroll
        for (int j = 0; j < 8; ++j)
            cs[j] = wmma_bf16(a, load_b<34>(Bt, wcol * 128 + j * 16, 0), cs[j]);
        __syncthreads();
    }

#pragma unroll
    for (int j = 0; j < 8; ++j) {
        int col = wcol * 128 + j * 16 + nl;
#pragma unroll
        for (int r = 0; r < 8; ++r) {
            int row = m0 + r + 8 * hf;
            int mg = m_base + row;
            float resid = bf2f(x1b[(size_t)mg * EMB + col]);
            Rs[row * 256 + col] = cs[j][r] + b2f[col] + resid;
        }
    }
    __syncthreads();

    for (int rr = wave * 8; rr < wave * 8 + 8; ++rr) {
        int mg = m_base + rr;
        float s = 0.0f;
#pragma unroll
        for (int jj = 0; jj < 8; ++jj) s += Rs[rr * 256 + lane + jj * 32];
        s += __shfl_xor(s, 1);  s += __shfl_xor(s, 2);  s += __shfl_xor(s, 4);
        s += __shfl_xor(s, 8);  s += __shfl_xor(s, 16);
        float mu = s * (1.0f / 256.0f);
        float vs = 0.0f;
#pragma unroll
        for (int jj = 0; jj < 8; ++jj) {
            float d = Rs[rr * 256 + lane + jj * 32] - mu;
            vs += d * d;
        }
        vs += __shfl_xor(vs, 1);  vs += __shfl_xor(vs, 2);  vs += __shfl_xor(vs, 4);
        vs += __shfl_xor(vs, 8);  vs += __shfl_xor(vs, 16);
        float inv = rsqrtf(vs * (1.0f / 256.0f) + 1e-5f);
#pragma unroll
        for (int jj = 0; jj < 8; ++jj) {
            int col = lane + jj * 32;
            float y = (Rs[rr * 256 + col] - mu) * inv * g2[col] + bb2[col];
            out[(size_t)mg * EMB + col] = y;
        }
    }
}

// =====================================================================
extern "C" void kernel_launch(void* const* d_in, const int* in_sizes, int n_in,
                              void* d_out, int out_size, void* d_ws, size_t ws_size,
                              hipStream_t stream) {
    const float* left  = (const float*)d_in[0];
    const float* right = (const float*)d_in[1];
    const float* qw    = (const float*)d_in[2];
    const float* kw    = (const float*)d_in[3];
    const float* vw    = (const float*)d_in[4];
    const float* pw    = (const float*)d_in[5];
    const float* pb    = (const float*)d_in[6];
    const float* ln1g  = (const float*)d_in[7];
    const float* ln1b  = (const float*)d_in[8];
    const float* ln2g  = (const float*)d_in[9];
    const float* ln2b  = (const float*)d_in[10];
    const float* w1    = (const float*)d_in[11];
    const float* b1    = (const float*)d_in[12];
    const float* w2    = (const float*)d_in[13];
    const float* b2    = (const float*)d_in[14];

    char* ws = (char*)d_ws;
    const size_t SZ = (size_t)BB * HN * HW * HS * 2;   // 64 MB
    __bf16* qb   = (__bf16*)(ws);
    __bf16* kb   = (__bf16*)(ws + SZ);
    __bf16* vb   = (__bf16*)(ws + 2 * SZ);
    __bf16* ob   = (__bf16*)(ws + 3 * SZ);
    __bf16* hbuf = (__bf16*)(ws);            // reuses q/k/v/O region (dead by then)
    __bf16* x1b  = (__bf16*)(ws + 4 * SZ);   // 64 MB

    hipLaunchKernelGGL(qkv_rope_kernel, dim3(HW / 128, BB * HN, 3), dim3(256), 0, stream,
                       left, right, qw, kw, vw, qb, kb, vb);
    hipLaunchKernelGGL(attention_kernel, dim3(BB * HH * HN), dim3(256), 0, stream,
                       qb, kb, vb, ob);
    hipLaunchKernelGGL(proj_ln_kernel, dim3((BB * HH * WW) / 64), dim3(256), 0, stream,
                       ob, pw, pb, left, ln1g, ln1b, x1b);
    hipLaunchKernelGGL(ffn1_kernel, dim3((BB * HH * WW) / 64, FF / 128), dim3(256), 0, stream,
                       x1b, w1, b1, hbuf);
    hipLaunchKernelGGL(ffn2_ln_kernel, dim3((BB * HH * WW) / 64), dim3(256), 0, stream,
                       hbuf, w2, b2, x1b, ln2g, ln2b, (float*)d_out);
}
